// Block_43447889166825
// MI455X (gfx1250) — compile-verified, matrix operand order checked by reading, not run on hardware
//
#include <hip/hip_runtime.h>
#include <cstddef>
#include <cstdint>

// ---------------------------------------------------------------------------
// Types for CDNA5 WMMA (wave32): 16x16x32 bf16 -> f32
// ---------------------------------------------------------------------------
typedef __attribute__((ext_vector_type(16))) __bf16 v16bf;
typedef __attribute__((ext_vector_type(8)))  __bf16 v8bf;
typedef __attribute__((ext_vector_type(8)))  float  v8f;

#define D_MODEL 2048
#define N_HEADS 16
#define HEAD_DIM 128
#define HIDDEN 8192
#define BATCH 2
#define SEQ 2048
#define MROWS (BATCH * SEQ)   // 4096

static __device__ inline __bf16 f2bf(float f) {
  union { float f; unsigned u; } a; a.f = f;
  unsigned r = a.u + 0x7FFFu + ((a.u >> 16) & 1u);   // round-nearest-even
  unsigned short hb = (unsigned short)(r >> 16);
  __bf16 out; __builtin_memcpy(&out, &hb, 2); return out;
}
static __device__ inline float bf2f(__bf16 b) {
  unsigned short hb; __builtin_memcpy(&hb, &b, 2);
  union { unsigned u; float f; } a; a.u = ((unsigned)hb) << 16;
  return a.f;
}

// WMMA operand fragment from row-major [rows][ld] bf16 tile (ISA 7.12.2):
// lane l holds row l%16; element i maps to k = (i/8)*16 + (l/16)*8 + i%8.
static __device__ inline v16bf ldfrag(const __bf16* p, int khalf) {
  v8bf lo = *(const v8bf*)(p + khalf * 8);
  v8bf hi = *(const v8bf*)(p + 16 + khalf * 8);
  v16bf f;
#pragma unroll
  for (int i = 0; i < 8; ++i) { f[i] = lo[i]; f[8 + i] = hi[i]; }
  return f;
}

// Async global -> LDS copy, 16B per lane (ASYNCcnt-tracked, ISA ch.10 / §15.18).
// GV addressing: LDS byte address in the vdst VGPR, global address in the 64-bit vaddr pair.
static __device__ inline void async_ld_b128(unsigned lds_addr, const void* gptr) {
  asm volatile("global_load_async_to_lds_b128 %0, %1, off"
               :: "v"(lds_addr), "v"(gptr)
               : "memory");
}
static __device__ inline unsigned lds_off(const void* p) {
  // Generic LDS addresses truncate to the wave-relative LDS byte offset.
  return (unsigned)(size_t)p;
}

// ---------------------------------------------------------------------------
// fp32 -> bf16 cast (weights / activations)
// ---------------------------------------------------------------------------
__global__ __launch_bounds__(256) void cast_f32_bf16(const float* __restrict__ in,
                                                     __bf16* __restrict__ out, size_t n) {
  size_t i = (size_t)blockIdx.x * 256 + threadIdx.x;
  if (i < n) out[i] = f2bf(in[i]);
}

// ---------------------------------------------------------------------------
// RMSNorm: one block per row, out = bf16(x * w * rsqrt(mean(x^2)+eps))
// ---------------------------------------------------------------------------
__global__ __launch_bounds__(256) void rmsnorm_kernel(const float* __restrict__ x,
                                                      const float* __restrict__ w,
                                                      __bf16* __restrict__ out) {
  const int row = blockIdx.x;
  const float* xr = x + (size_t)row * D_MODEL;
  float s = 0.f;
  for (int i = threadIdx.x; i < D_MODEL; i += 256) { float v = xr[i]; s += v * v; }
  __shared__ float red[256];
  red[threadIdx.x] = s;
  __syncthreads();
  for (int off = 128; off > 0; off >>= 1) {
    if (threadIdx.x < off) red[threadIdx.x] += red[threadIdx.x + off];
    __syncthreads();
  }
  float rs = rsqrtf(red[0] * (1.0f / D_MODEL) + 1e-6f);
  for (int i = threadIdx.x; i < D_MODEL; i += 256)
    out[(size_t)row * D_MODEL + i] = f2bf(xr[i] * w[i] * rs);
}

// ---------------------------------------------------------------------------
// GEMM: C[M,N] = A[M,K](bf16) @ W[N,K]^T(bf16)  (+bias[col]) (+res[row,col])
// 256 thr = 8 waves. Block tile 128x128, BK=64, double-buffered LDS filled by
// async global->LDS loads. Wave grid 4x2, each wave owns 32x64 (8 f32 accs).
// ---------------------------------------------------------------------------
template <bool OUTBF, bool RES, bool BIAS>
__global__ __launch_bounds__(256) void gemm_bf16_kernel(
    const __bf16* __restrict__ A, const __bf16* __restrict__ W,
    float* __restrict__ Cf, __bf16* __restrict__ Cb,
    const float* __restrict__ res, const float* __restrict__ bias,
    int Mtot, int N, int K) {
  __shared__ __align__(16) __bf16 Abuf[2][128 * 64];
  __shared__ __align__(16) __bf16 Bbuf[2][128 * 64];

  const int bn = blockIdx.x, bm = blockIdx.y;
  const int t = threadIdx.x;
  const int wid = t >> 5, lane = t & 31;
  const int mrow = lane & 15, khalf = lane >> 4;
  const int wr = wid & 3;   // 4 row-waves * 32 = 128 rows
  const int wc = wid >> 2;  // 2 col-waves * 64 = 128 cols

  const unsigned aL = lds_off(&Abuf[0][0]);
  const unsigned bL = lds_off(&Bbuf[0][0]);

  // per-thread copy chunks: 1024 x 16B per tile; 4 A + 4 B chunks per thread
  auto issue_tile = [&](int kt, int buf) {
    const int k0 = kt * 64;
#pragma unroll
    for (int j = 0; j < 4; ++j) {
      int c = t + 256 * j;            // 0..1023
      int r = c >> 3;                 // 0..127
      int ko = (c & 7) * 8;           // 0..56
      unsigned loff = (unsigned)((r * 64 + ko) * 2);
      async_ld_b128(aL + (unsigned)buf * 16384u + loff,
                    A + (size_t)(bm * 128 + r) * K + k0 + ko);
      async_ld_b128(bL + (unsigned)buf * 16384u + loff,
                    W + (size_t)(bn * 128 + r) * K + k0 + ko);
    }
  };

  v8f acc[2][4] = {};

  const int NT = K >> 6;
  issue_tile(0, 0);
  for (int i = 0; i < NT; ++i) {
    const int cur = i & 1;
    if (i + 1 < NT) {
      issue_tile(i + 1, cur ^ 1);
      asm volatile("s_wait_asynccnt 8" ::: "memory");  // previous tile resident
    } else {
      asm volatile("s_wait_asynccnt 0" ::: "memory");
    }
    __syncthreads();

    const __bf16* At = Abuf[cur];
    const __bf16* Bt = Bbuf[cur];
#pragma unroll
    for (int ks = 0; ks < 64; ks += 32) {
      v16bf a0 = ldfrag(At + (wr * 32 + mrow) * 64 + ks, khalf);
      v16bf a1 = ldfrag(At + (wr * 32 + 16 + mrow) * 64 + ks, khalf);
#pragma unroll
      for (int nt = 0; nt < 4; ++nt) {
        v16bf bb = ldfrag(Bt + (wc * 64 + nt * 16 + mrow) * 64 + ks, khalf);
        acc[0][nt] = __builtin_amdgcn_wmma_f32_16x16x32_bf16(false, a0, false, bb, (short)0, acc[0][nt], false, false);
        acc[1][nt] = __builtin_amdgcn_wmma_f32_16x16x32_bf16(false, a1, false, bb, (short)0, acc[1][nt], false, false);
      }
    }
    __syncthreads();  // all waves done with buf[cur] before it is refilled
  }

  // C layout: lane l holds (row = v + 8*(l/16), col = l%16) for v in 0..7
#pragma unroll
  for (int mt = 0; mt < 2; ++mt) {
#pragma unroll
    for (int nt = 0; nt < 4; ++nt) {
#pragma unroll
      for (int v = 0; v < 8; ++v) {
        int row = bm * 128 + wr * 32 + mt * 16 + v + 8 * khalf;
        int col = bn * 128 + wc * 64 + nt * 16 + mrow;
        float val = acc[mt][nt][v];
        if (BIAS) val += bias[col];
        if (RES)  val += res[(size_t)row * N + col];
        if (OUTBF) Cb[(size_t)row * N + col] = f2bf(val);
        else       Cf[(size_t)row * N + col] = val;
      }
    }
  }
}

// ---------------------------------------------------------------------------
// RoPE (interleaved-pair) applied in-place to q and k: [B, L, H, 128] bf16
// ---------------------------------------------------------------------------
__global__ __launch_bounds__(256) void rope_kernel(__bf16* __restrict__ q, __bf16* __restrict__ k,
                                                   const float* __restrict__ cosv,
                                                   const float* __restrict__ sinv, int n) {
  int i = blockIdx.x * 256 + threadIdx.x;  // B*L*H*64 pairs
  if (i >= n) return;
  int p = i & 63;
  int h = (i >> 6) & 15;
  int l = (i >> 10) & 2047;
  int b = i >> 21;
  size_t base = ((((size_t)b * SEQ + l) * N_HEADS + h) * HEAD_DIM) + 2 * p;
  float c = cosv[l * 64 + p], s = sinv[l * 64 + p];
  float q0 = bf2f(q[base]), q1 = bf2f(q[base + 1]);
  q[base]     = f2bf(q0 * c - q1 * s);
  q[base + 1] = f2bf(q0 * s + q1 * c);
  float k0 = bf2f(k[base]), k1 = bf2f(k[base + 1]);
  k[base]     = f2bf(k0 * c - k1 * s);
  k[base + 1] = f2bf(k0 * s + k1 * c);
}

// ---------------------------------------------------------------------------
// Flash attention (causal): 128 thr (4 waves), Q tile 64 rows, KV tile 32.
// QK^T and PV via WMMA bf16; online softmax via 16-lane shfl_xor reductions
// (matches WMMA C-layout: each 16-lane half owns 8 output rows).
// ---------------------------------------------------------------------------
__global__ __launch_bounds__(128) void attn_kernel(const __bf16* __restrict__ q,
                                                   const __bf16* __restrict__ k,
                                                   const __bf16* __restrict__ v,
                                                   __bf16* __restrict__ o) {
  const int qt = blockIdx.x;
  const int h  = blockIdx.y;
  const int b  = blockIdx.z;
  const int t = threadIdx.x;
  const int wid = t >> 5, lane = t & 31;
  const int mrow = lane & 15, khalf = lane >> 4;

  __shared__ __align__(16) __bf16 Qs[64 * 128];
  __shared__ __align__(16) __bf16 Ks[32 * 128];
  __shared__ __align__(16) __bf16 Vt[128 * 32];   // transposed: [d][kv]
  __shared__ __align__(16) __bf16 Ps[4][16 * 32]; // per-wave P tile

  const int q0 = qt * 64;

  for (int i = t; i < 1024; i += 128) {           // Q tile 64x128
    int r = i >> 4, c = (i & 15) * 8;
    ((uint4*)Qs)[i] =
        *(const uint4*)(q + (((size_t)b * SEQ + q0 + r) * N_HEADS + h) * HEAD_DIM + c);
  }
  __syncthreads();

  float m[8], lsum[8];
#pragma unroll
  for (int i = 0; i < 8; ++i) { m[i] = -3.0e38f; lsum[i] = 0.f; }
  v8f O[8] = {};

  const float scale = 0.08838834764831845f;  // 1/sqrt(128)
  const int qrow_base = q0 + wid * 16;
  const int nkt = (q0 + 64) / 32;

  for (int kt = 0; kt < nkt; ++kt) {
    const int k0 = kt * 32;
    for (int i = t; i < 512; i += 128) {           // K tile 32x128
      int r = i >> 4, c = (i & 15) * 8;
      ((uint4*)Ks)[i] =
          *(const uint4*)(k + (((size_t)b * SEQ + k0 + r) * N_HEADS + h) * HEAD_DIM + c);
    }
    for (int i = t; i < 4096; i += 128) {          // V transpose into [d][kv]
      int r = i >> 7, c = i & 127;
      Vt[c * 32 + r] = v[(((size_t)b * SEQ + k0 + r) * N_HEADS + h) * HEAD_DIM + c];
    }
    __syncthreads();

    v8f s0 = {}; v8f s1 = {};
#pragma unroll
    for (int d0 = 0; d0 < 128; d0 += 32) {
      v16bf aQ = ldfrag(Qs + (wid * 16 + mrow) * 128 + d0, khalf);
      v16bf bA = ldfrag(Ks + mrow * 128 + d0, khalf);
      v16bf bB = ldfrag(Ks + (16 + mrow) * 128 + d0, khalf);
      s0 = __builtin_amdgcn_wmma_f32_16x16x32_bf16(false, aQ, false, bA, (short)0, s0, false, false);
      s1 = __builtin_amdgcn_wmma_f32_16x16x32_bf16(false, aQ, false, bB, (short)0, s1, false, false);
    }

#pragma unroll
    for (int vv = 0; vv < 8; ++vv) {
      int qg = qrow_base + vv + 8 * khalf;
      int kc0 = k0 + mrow;
      int kc1 = k0 + 16 + mrow;
      float x0 = (kc0 <= qg) ? s0[vv] * scale : -1e30f;
      float x1 = (kc1 <= qg) ? s1[vv] * scale : -1e30f;
      float mv = fmaxf(x0, x1);
      mv = fmaxf(mv, __shfl_xor(mv, 1));
      mv = fmaxf(mv, __shfl_xor(mv, 2));
      mv = fmaxf(mv, __shfl_xor(mv, 4));
      mv = fmaxf(mv, __shfl_xor(mv, 8));
      float mnew = fmaxf(m[vv], mv);
      float alpha = __expf(m[vv] - mnew);
      float p0 = __expf(x0 - mnew);
      float p1 = __expf(x1 - mnew);
      float ps = p0 + p1;
      ps += __shfl_xor(ps, 1);
      ps += __shfl_xor(ps, 2);
      ps += __shfl_xor(ps, 4);
      ps += __shfl_xor(ps, 8);
      lsum[vv] = lsum[vv] * alpha + ps;
      m[vv] = mnew;
      Ps[wid][(vv + 8 * khalf) * 32 + mrow]      = f2bf(p0);
      Ps[wid][(vv + 8 * khalf) * 32 + 16 + mrow] = f2bf(p1);
#pragma unroll
      for (int nt = 0; nt < 8; ++nt) O[nt][vv] *= alpha;
    }

    {
      v16bf aP = ldfrag(Ps[wid] + mrow * 32, khalf);
#pragma unroll
      for (int nt = 0; nt < 8; ++nt) {
        v16bf bV = ldfrag(Vt + (nt * 16 + mrow) * 32, khalf);
        O[nt] = __builtin_amdgcn_wmma_f32_16x16x32_bf16(false, aP, false, bV, (short)0, O[nt], false, false);
      }
    }
    __syncthreads();
  }

#pragma unroll
  for (int vv = 0; vv < 8; ++vv) {
    float inv = 1.0f / lsum[vv];
    int qg = qrow_base + vv + 8 * khalf;
#pragma unroll
    for (int nt = 0; nt < 8; ++nt) {
      o[(((size_t)b * SEQ + qg) * N_HEADS + h) * HEAD_DIM + nt * 16 + mrow] =
          f2bf(O[nt][vv] * inv);
    }
  }
}

// ---------------------------------------------------------------------------
// SwiGLU gate: g <- silu(g) * t  (in place on g, bf16)
// ---------------------------------------------------------------------------
__global__ __launch_bounds__(256) void silumul_kernel(__bf16* __restrict__ g,
                                                      const __bf16* __restrict__ t2, size_t n) {
  size_t i = (size_t)blockIdx.x * 256 + threadIdx.x;
  if (i >= n) return;
  float gv = bf2f(g[i]);
  float tv = bf2f(t2[i]);
  float s = gv / (1.f + __expf(-gv));
  g[i] = f2bf(s * tv);
}

// ---------------------------------------------------------------------------
// Host launcher
// ---------------------------------------------------------------------------
extern "C" void kernel_launch(void* const* d_in, const int* in_sizes, int n_in,
                              void* d_out, int out_size, void* d_ws, size_t ws_size,
                              hipStream_t stream) {
  (void)in_sizes; (void)n_in; (void)out_size; (void)ws_size;
  const float* x    = (const float*)d_in[0];
  // d_in[1] = attention_mask (pure causal; implemented analytically)
  const float* cosv = (const float*)d_in[2];
  const float* sinv = (const float*)d_in[3];
  const float* anw  = (const float*)d_in[4];
  const float* Wq   = (const float*)d_in[5];
  const float* Wk   = (const float*)d_in[6];
  const float* Wv   = (const float*)d_in[7];
  const float* Wo   = (const float*)d_in[8];
  const float* Wob  = (const float*)d_in[9];
  const float* fnw  = (const float*)d_in[10];
  const float* Wg   = (const float*)d_in[11];
  const float* Wh   = (const float*)d_in[12];
  const float* Wout = (const float*)d_in[13];
  float* out = (float*)d_out;

  char* ws = (char*)d_ws;
  size_t off = 0;
  auto alloc = [&](size_t bytes) -> void* {
    void* p = ws + off;
    off += (bytes + 255) & ~(size_t)255;
    return p;
  };
  const size_t DD = (size_t)D_MODEL * D_MODEL;
  const size_t DH = (size_t)D_MODEL * HIDDEN;
  const size_t MD = (size_t)MROWS * D_MODEL;
  const size_t MH = (size_t)MROWS * HIDDEN;

  __bf16* Wq_b  = (__bf16*)alloc(DD * 2);
  __bf16* Wk_b  = (__bf16*)alloc(DD * 2);
  __bf16* Wv_b  = (__bf16*)alloc(DD * 2);
  __bf16* Wo_b  = (__bf16*)alloc(DD * 2);
  __bf16* Wg_b  = (__bf16*)alloc(DH * 2);
  __bf16* Wh_b  = (__bf16*)alloc(DH * 2);
  __bf16* Wou_b = (__bf16*)alloc(DH * 2);
  __bf16* h_b   = (__bf16*)alloc(MD * 2);
  __bf16* q_b   = (__bf16*)alloc(MD * 2);
  __bf16* k_b   = (__bf16*)alloc(MD * 2);
  __bf16* v_b   = (__bf16*)alloc(MD * 2);
  __bf16* o_b   = (__bf16*)alloc(MD * 2);
  float*  x1    = (float*)alloc(MD * 4);
  __bf16* h2_b  = (__bf16*)alloc(MD * 2);
  __bf16* g_b   = (__bf16*)alloc(MH * 2);
  __bf16* t_b   = (__bf16*)alloc(MH * 2);

  auto cast = [&](const float* src, __bf16* dst, size_t n) {
    cast_f32_bf16<<<(unsigned)((n + 255) / 256), 256, 0, stream>>>(src, dst, n);
  };
  cast(Wq, Wq_b, DD);
  cast(Wk, Wk_b, DD);
  cast(Wv, Wv_b, DD);
  cast(Wo, Wo_b, DD);
  cast(Wg, Wg_b, DH);
  cast(Wh, Wh_b, DH);
  cast(Wout, Wou_b, DH);

  // h = rmsnorm(x) * attn_norm_w
  rmsnorm_kernel<<<MROWS, 256, 0, stream>>>(x, anw, h_b);

  // q,k,v = h @ W{q,k,v}^T
  dim3 gQKV(D_MODEL / 128, MROWS / 128);
  gemm_bf16_kernel<true, false, false><<<gQKV, 256, 0, stream>>>(
      h_b, Wq_b, nullptr, q_b, nullptr, nullptr, MROWS, D_MODEL, D_MODEL);
  gemm_bf16_kernel<true, false, false><<<gQKV, 256, 0, stream>>>(
      h_b, Wk_b, nullptr, k_b, nullptr, nullptr, MROWS, D_MODEL, D_MODEL);
  gemm_bf16_kernel<true, false, false><<<gQKV, 256, 0, stream>>>(
      h_b, Wv_b, nullptr, v_b, nullptr, nullptr, MROWS, D_MODEL, D_MODEL);

  // RoPE in place on q,k
  const int npairs = BATCH * SEQ * N_HEADS * (HEAD_DIM / 2);
  rope_kernel<<<npairs / 256, 256, 0, stream>>>(q_b, k_b, cosv, sinv, npairs);

  // Causal flash attention -> o
  attn_kernel<<<dim3(SEQ / 64, N_HEADS, BATCH), 128, 0, stream>>>(q_b, k_b, v_b, o_b);

  // x1 = x + o @ Wo^T + Wo_b
  gemm_bf16_kernel<false, true, true><<<gQKV, 256, 0, stream>>>(
      o_b, Wo_b, x1, nullptr, x, Wob, MROWS, D_MODEL, D_MODEL);

  // h2 = rmsnorm(x1) * ffn_norm_w
  rmsnorm_kernel<<<MROWS, 256, 0, stream>>>(x1, fnw, h2_b);

  // g = h2 @ Wg^T ; t = h2 @ Wh^T
  dim3 gFFN(HIDDEN / 128, MROWS / 128);
  gemm_bf16_kernel<true, false, false><<<gFFN, 256, 0, stream>>>(
      h2_b, Wg_b, nullptr, g_b, nullptr, nullptr, MROWS, HIDDEN, D_MODEL);
  gemm_bf16_kernel<true, false, false><<<gFFN, 256, 0, stream>>>(
      h2_b, Wh_b, nullptr, t_b, nullptr, nullptr, MROWS, HIDDEN, D_MODEL);

  // g <- silu(g) * t
  silumul_kernel<<<(unsigned)(MH / 256), 256, 0, stream>>>(g_b, t_b, MH);

  // out = x1 + g @ Wout^T
  gemm_bf16_kernel<false, true, false><<<gQKV, 256, 0, stream>>>(
      g_b, Wou_b, out, nullptr, x1, nullptr, MROWS, D_MODEL, HIDDEN);
}
